// MyLSTMCell_75634374082645
// MI455X (gfx1250) — matrix operand order repaired and checked by reading, not compiled, not run on hardware
//
#include <hip/hip_runtime.h>
#include <cstddef>

typedef __attribute__((ext_vector_type(16))) __bf16 v16bf;
typedef __attribute__((ext_vector_type(8)))  __bf16 v8bf;
typedef __attribute__((ext_vector_type(4)))  __bf16 v4bf;
typedef __attribute__((ext_vector_type(8)))  float  v8f;
typedef __attribute__((ext_vector_type(4)))  float  v4f;
typedef __attribute__((ext_vector_type(4)))  int    v4i;

#define BM 64      // rows per block
#define BH 64      // h-columns per block (per gate)
#define BK 32      // K chunk
#define APITCH 40  // LDS row pitch in bf16 elems (80B, multiple of 16B)
#define WPITCH 40
#define NTHREADS 256

#if defined(__has_builtin)
#if __has_builtin(__builtin_amdgcn_global_load_async_to_lds_b128) && \
    __has_builtin(__builtin_amdgcn_s_wait_asynccnt)
#define ASYNC_COPY 1
#endif
#endif
#ifndef ASYNC_COPY
#define ASYNC_COPY 0
#endif

__device__ __forceinline__ float sigmoidf_(float x) { return 1.f / (1.f + __expf(-x)); }
__device__ __forceinline__ float tanh_fast(float x) {
  x = fminf(fmaxf(x, -10.f), 10.f);
  float e = __expf(2.f * x);
  return (e - 1.f) / (e + 1.f);
}

// ============================================================
// Kernel 1: pack fused weight [Wx;Wh] (f32, [2048][4096] K-major)
// into bf16 N-major Wp[4096][2048] via tiled LDS transpose.
// ============================================================
__global__ __launch_bounds__(NTHREADS)
void pack_weights(const float* __restrict__ Wx, const float* __restrict__ Wh,
                  __bf16* __restrict__ Wp)
{
  __shared__ __align__(16) __bf16 T[64][72];   // pitch 144B (multiple of 16B)

  const int tid = threadIdx.x;
  const int k0  = blockIdx.x * 64;   // 0..2047, never straddles 1024
  const int n0  = blockIdx.y * 64;   // 0..4095

  const float* src = (k0 < 1024) ? (Wx + (size_t)k0 * 4096)
                                 : (Wh + (size_t)(k0 - 1024) * 4096);
  #pragma unroll
  for (int it = 0; it < 4; ++it) {
    int q    = tid + it * NTHREADS;     // 0..1023
    int krow = q >> 4;                  // 16 float4 per k-row
    int n4   = q & 15;
    v4f w = *(const v4f*)(src + (size_t)krow * 4096 + n0 + n4 * 4);
    T[n4 * 4 + 0][krow] = (__bf16)w.x;
    T[n4 * 4 + 1][krow] = (__bf16)w.y;
    T[n4 * 4 + 2][krow] = (__bf16)w.z;
    T[n4 * 4 + 3][krow] = (__bf16)w.w;
  }
  __syncthreads();
  #pragma unroll
  for (int it = 0; it < 2; ++it) {
    int q  = tid + it * NTHREADS;       // 0..511
    int n  = q >> 3;                    // 8 groups of 8 bf16 per row
    int kg = q & 7;
    v8bf v = *(const v8bf*)&T[n][kg * 8];
    *(v8bf*)(Wp + (size_t)(n0 + n) * 2048 + k0 + kg * 8) = v;
  }
}

// ---- A tile: global f32 -> registers ----
__device__ __forceinline__ void stage_A_global(
    const float* __restrict__ x, const float* __restrict__ state,
    int m0, int kc, int tid, v4f (&ar)[2])
{
  const float* Asrc = (kc < 1024) ? (x + kc) : (state + (kc - 1024));
  #pragma unroll
  for (int it = 0; it < 2; ++it) {               // 64 rows x 32 f32
    int q   = tid + it * NTHREADS;               // 0..511
    int row = q >> 3;                            // 8 float4 per row
    int c4  = q & 7;
    ar[it] = *(const v4f*)(Asrc + (size_t)(m0 + row) * 1024 + c4 * 4);
  }
}

// ---- A tile: registers -> LDS via hw bf16 convert ----
__device__ __forceinline__ void stage_A_lds(__bf16* Asb, int tid, const v4f (&ar)[2])
{
  #pragma unroll
  for (int it = 0; it < 2; ++it) {
    int q = tid + it * NTHREADS;
    int row = q >> 3, c4 = q & 7;
    v4bf pk = { (__bf16)ar[it].x, (__bf16)ar[it].y, (__bf16)ar[it].z, (__bf16)ar[it].w };
    *(v4bf*)(Asb + row * APITCH + c4 * 4) = pk;   // ds_store_b64
  }
}

// ---- W tiles: packed bf16 global -> LDS (async copy, no VGPR round-trip) ----
__device__ __forceinline__ void stage_W(const __bf16* __restrict__ Wp, __bf16* Wsb,
                                        int h0, int kc, int tid)
{
  #pragma unroll
  for (int it = 0; it < 4; ++it) {               // 256 n-rows x 32 bf16
    int q   = tid + it * NTHREADS;               // 0..1023
    int nl  = q >> 2;                            // gate*64 + n
    int k8  = q & 3;
    int row = (nl >> 6) * 1024 + h0 + (nl & 63);
    const __bf16* g = Wp + (size_t)row * 2048 + kc + k8 * 8;
    __bf16*       l = Wsb + (size_t)nl * WPITCH + k8 * 8;
#if ASYNC_COPY
    __builtin_amdgcn_global_load_async_to_lds_b128(
        (__attribute__((address_space(1))) v4i*)g,
        (__attribute__((address_space(3))) v4i*)l, 0, 0);
#else
    *(v8bf*)l = *(const v8bf*)g;
#endif
  }
}

__device__ __forceinline__ void wait_async_copies()
{
#if ASYNC_COPY
  __builtin_amdgcn_s_wait_asynccnt(0);
#endif
}

// ============================================================
// Kernel 2: fused LSTM-cell GEMM + gate epilogue
// ============================================================
__global__ __launch_bounds__(NTHREADS, 1)
void lstm_cell_wmma_bf16(const float* __restrict__ x, const float* __restrict__ state,
                         const float* __restrict__ cell, const __bf16* __restrict__ Wp,
                         const float* __restrict__ bx, const float* __restrict__ bh,
                         float* __restrict__ out)
{
  __shared__ __align__(16) __bf16 As[2][BM][APITCH];        // 10 KB
  __shared__ __align__(16) __bf16 Ws[2][4][BH][WPITCH];     // 40 KB

  const int tid    = threadIdx.x;
  const int lane   = tid & 31;
  const int wave   = tid >> 5;
  const int lane16 = lane & 15;
  const int laneHi = lane >> 4;   // half-wave select
  const int waveM  = wave & 1;    // 2 M groups of 32 rows
  const int waveH  = wave >> 1;   // 4 H groups of 16 cols

  const int h0 = blockIdx.x * BH;   // h tile within [0,1024)
  const int m0 = blockIdx.y * BM;   // row tile within [0,16384)

  v8f acc[4][2];                    // [gate][m-subtile], 64 VGPRs
  #pragma unroll
  for (int g = 0; g < 4; ++g)
    #pragma unroll
    for (int mi = 0; mi < 2; ++mi)
      #pragma unroll
      for (int r = 0; r < 8; ++r) acc[g][mi][r] = 0.f;

  v4f ar[2];

  // prologue: chunk 0 -> LDS buffer 0
  stage_A_global(x, state, m0, 0, tid, ar);
  stage_W(Wp, &Ws[0][0][0][0], h0, 0, tid);
  stage_A_lds(&As[0][0][0], tid, ar);
  wait_async_copies();
  __syncthreads();

  const int NCHUNK = 2048 / BK;     // 64
  #pragma clang loop unroll(disable)
  for (int ci = 0; ci < NCHUNK; ++ci) {
    const int  buf = ci & 1;
    const bool nxt = (ci + 1) < NCHUNK;
    if (nxt) {
      stage_A_global(x, state, m0, (ci + 1) * BK, tid, ar);
      stage_W(Wp, &Ws[buf ^ 1][0][0][0], h0, (ci + 1) * BK, tid);
    }

    // ---- compute on current buffer ----
    const __bf16* Asb = &As[buf][0][0];
    const __bf16* Wsb = &Ws[buf][0][0][0];

    v16bf afrag[2];
    #pragma unroll
    for (int mi = 0; mi < 2; ++mi) {
      int row = waveM * 32 + mi * 16 + lane16;
      int kb  = laneHi ? 8 : 0;     // ISA A layout: K 0-7/16-23 vs 8-15/24-31
      v8bf lo = *(const v8bf*)(Asb + row * APITCH + kb);
      v8bf hi = *(const v8bf*)(Asb + row * APITCH + kb + 16);
      afrag[mi] = __builtin_shufflevector(lo, hi, 0,1,2,3,4,5,6,7,8,9,10,11,12,13,14,15);
    }
    #pragma unroll
    for (int g = 0; g < 4; ++g) {
      int n  = waveH * 16 + lane16;
      int ko = laneHi ? 8 : 0;      // B layout matches A: K 0-7/16-23 vs 8-15/24-31
      const __bf16* bp = Wsb + (size_t)(g * BH + n) * WPITCH + ko;
      v8bf lo = *(const v8bf*)bp;
      v8bf hi = *(const v8bf*)(bp + 16);
      v16bf bfrag = __builtin_shufflevector(lo, hi, 0,1,2,3,4,5,6,7,8,9,10,11,12,13,14,15);
      acc[g][0] = __builtin_amdgcn_wmma_f32_16x16x32_bf16(
          false, afrag[0], false, bfrag, (short)0, acc[g][0], false, false);
      acc[g][1] = __builtin_amdgcn_wmma_f32_16x16x32_bf16(
          false, afrag[1], false, bfrag, (short)0, acc[g][1], false, false);
    }

    if (nxt) stage_A_lds(&As[buf ^ 1][0][0], tid, ar);
    wait_async_copies();
    __syncthreads();
  }

  // ---- fused LSTM epilogue ----
  const int hcol = h0 + waveH * 16 + lane16;
  float bias[4];
  #pragma unroll
  for (int g = 0; g < 4; ++g)
    bias[g] = bx[g * 1024 + hcol] + bh[g * 1024 + hcol];

  float* outS = out;                              // statedot
  float* outC = out + (size_t)16384 * 1024;       // celldot
  #pragma unroll
  for (int mi = 0; mi < 2; ++mi) {
    #pragma unroll
    for (int r = 0; r < 8; ++r) {
      int m = m0 + waveM * 32 + mi * 16 + laneHi * 8 + r;  // C layout: half-wave = +8 rows
      size_t idx = (size_t)m * 1024 + hcol;
      float gi = acc[0][mi][r] + bias[0];
      float gj = acc[1][mi][r] + bias[1];
      float gf = acc[2][mi][r] + bias[2];
      float go = acc[3][mi][r] + bias[3];
      float i_ = tanh_fast(gi);
      float j_ = sigmoidf_(gj);
      float f_ = sigmoidf_(gf);
      float o_ = tanh_fast(go);
      float cd = cell[idx] * f_ + i_ * j_;
      float sd = tanh_fast(cd) * o_;
      outS[idx] = sd;
      outC[idx] = cd;
    }
  }
}

extern "C" void kernel_launch(void* const* d_in, const int* in_sizes, int n_in,
                              void* d_out, int out_size, void* d_ws, size_t ws_size,
                              hipStream_t stream) {
  (void)in_sizes; (void)n_in; (void)out_size; (void)ws_size;
  const float* x     = (const float*)d_in[0];
  const float* state = (const float*)d_in[1];
  const float* cell  = (const float*)d_in[2];
  const float* Wx    = (const float*)d_in[3];
  const float* bx    = (const float*)d_in[4];
  const float* Wh    = (const float*)d_in[5];
  const float* bh    = (const float*)d_in[6];
  float* out  = (float*)d_out;
  __bf16* Wp  = (__bf16*)d_ws;   // 4096*2048*2 = 16 MB of scratch

  dim3 gridP(2048 / 64, 4096 / 64);   // (32, 64)
  pack_weights<<<gridP, NTHREADS, 0, stream>>>(Wx, Wh, Wp);

  dim3 grid(1024 / BH, 16384 / BM);   // (16, 256)
  lstm_cell_wmma_bf16<<<grid, NTHREADS, 0, stream>>>(x, state, cell, Wp, bx, bh, out);
}
